// ProteinFeatures_54357106098677
// MI455X (gfx1250) — compile-verified, matrix operand order checked by reading, not run on hardware
//
#include <hip/hip_runtime.h>
#include <hip/hip_bf16.h>
#include <math.h>

typedef __attribute__((ext_vector_type(16))) _Float16 v16h;
typedef __attribute__((ext_vector_type(8)))  _Float16 v8h;
typedef __attribute__((ext_vector_type(8)))  float    v8f;

#define B_DIM   4
#define L_DIM   1536
#define KNB     48
#define EIN     416      // 16 PE + 25*16 RBF
#define EIN_P   424      // padded LDS row stride (212 words; 212 mod 64 = 20 -> conflict-free A rows)
#define EOUT    128
#define EOUT_P  132      // padded LDS row stride (8*132 mod 64 = 32 -> lanes 16-31 disjoint banks)
#define NTILES  8        // 128/16
#define KTILES  13       // 416/32
#define MTILES  3        // 48/16
#define CAND    (L_DIM / 256)   // 6 KNN candidates per thread

// Pair atom codes: 0=N 1=Ca 2=C 3=O 4=Cb  (24 pairs; RBF block 0 is the
// masked/adjusted Ca-Ca KNN distance handled separately)
__constant__ int c_pairA[24] = {0,2,3,4,1,1,1,1,0,0,0,4,4,3,0,2,3,4,2,3,4,2,3,2};
__constant__ int c_pairB[24] = {0,2,3,4,0,2,3,4,2,3,4,2,3,2,1,1,1,1,0,0,0,4,4,3};

// ---------------------------------------------------------------------------
// Kernel 1: virtual Cb atom per residue
// ---------------------------------------------------------------------------
__global__ __launch_bounds__(256) void cb_kernel(const float* __restrict__ X,
                                                 float* __restrict__ Cb) {
    int t = blockIdx.x * 256 + threadIdx.x;
    if (t >= B_DIM * L_DIM) return;
    const float* p = X + (size_t)t * 12;
    float nx = p[0], ny = p[1], nz = p[2];
    float cax = p[3], cay = p[4], caz = p[5];
    float cx = p[6], cy = p[7], cz = p[8];
    float bx = cax - nx, by = cay - ny, bz = caz - nz;      // Ca - N
    float vx = cx - cax, vy = cy - cay, vz = cz - caz;      // C - Ca
    float ax = by * vz - bz * vy;
    float ay = bz * vx - bx * vz;
    float az = bx * vy - by * vx;
    Cb[(size_t)t * 3 + 0] = -0.58273431f * ax + 0.56802827f * bx - 0.54067466f * vx + cax;
    Cb[(size_t)t * 3 + 1] = -0.58273431f * ay + 0.56802827f * by - 0.54067466f * vy + cay;
    Cb[(size_t)t * 3 + 2] = -0.58273431f * az + 0.56802827f * bz - 0.54067466f * vz + caz;
}

// ---------------------------------------------------------------------------
// Kernel 2: convert W_edge [416,128] f32 -> f16 in WMMA B-fragment lane layout.
// Fragment (nt,kt): lane holds 16 halves; lane = N%16 (+16 for K-halves 8..15),
// halves 0..7 -> K = kbase+klo+0..7, halves 8..15 -> K = kbase+klo+16+0..7.
// ---------------------------------------------------------------------------
__global__ __launch_bounds__(256) void wconv_kernel(const float* __restrict__ W,
                                                    _Float16* __restrict__ Wf) {
    int t = blockIdx.x * 256 + threadIdx.x;
    if (t >= NTILES * KTILES * 32) return;
    int nt = t / (KTILES * 32);
    int r  = t - nt * (KTILES * 32);
    int kt = r >> 5;
    int lane = r & 31;
    int n   = nt * 16 + (lane & 15);
    int klo = (lane & 16) ? 8 : 0;
    int kb  = kt * 32 + klo;
    _Float16* dst = Wf + (size_t)t * 16;
    for (int h = 0; h < 8; ++h) {
        dst[h]     = (_Float16)W[(kb + h) * EOUT + n];
        dst[8 + h] = (_Float16)W[(kb + 16 + h) * EOUT + n];
    }
}

// ---------------------------------------------------------------------------
// Kernel 3: masked KNN, register-resident candidates (6 per thread).
// One 256-thread block per (b,i); matches lax.top_k(-D_adjust) semantics
// (ascending distance, lowest-index tie-break).
// ---------------------------------------------------------------------------
__global__ __launch_bounds__(256) void knn_kernel(const float* __restrict__ X,
                                                  const float* __restrict__ mask,
                                                  int* __restrict__ eidx,
                                                  float* __restrict__ dnb,
                                                  int* __restrict__ eidx_out) {
    __shared__ float pv[256];
    __shared__ int   pi[256];
    __shared__ float sDmax;
    __shared__ int   sSelJ;

    const int tid = threadIdx.x;
    const int bi  = blockIdx.x;          // b*L + i
    const int b   = bi / L_DIM;
    const size_t base = (size_t)b * L_DIM;

    const float cax = X[((size_t)bi * 4 + 1) * 3 + 0];
    const float cay = X[((size_t)bi * 4 + 1) * 3 + 1];
    const float caz = X[((size_t)bi * 4 + 1) * 3 + 2];
    const float mi  = mask[bi];

    float lv[CAND], lm2[CAND];
    float lmax = 0.0f;
    #pragma unroll
    for (int q = 0; q < CAND; ++q) {
        int j = q * 256 + tid;
        size_t gj = base + j;
        float dx = cax - X[(gj * 4 + 1) * 3 + 0];
        float dy = cay - X[(gj * 4 + 1) * 3 + 1];
        float dz = caz - X[(gj * 4 + 1) * 3 + 2];
        float m2 = mi * mask[gj];
        float D = m2 * sqrtf(dx * dx + dy * dy + dz * dz + 1e-6f);
        lv[q] = D; lm2[q] = m2;
        lmax = fmaxf(lmax, D);
    }
    pv[tid] = lmax;
    __syncthreads();
    if (tid < 32) {
        float m = pv[tid];
        for (int s = 1; s < 8; ++s) m = fmaxf(m, pv[tid + 32 * s]);
        for (int msk = 16; msk >= 1; msk >>= 1) m = fmaxf(m, __shfl_xor(m, msk, 32));
        if (tid == 0) sDmax = m;
    }
    __syncthreads();
    const float Dmax = sDmax;
    #pragma unroll
    for (int q = 0; q < CAND; ++q) lv[q] += (1.0f - lm2[q]) * Dmax;

    for (int k = 0; k < KNB; ++k) {
        float bv = __builtin_inff();
        int   bj = 0x7fffffff;
        #pragma unroll
        for (int q = 0; q < CAND; ++q) {     // j increases with q: strict < keeps lowest j on ties
            float v = lv[q];
            if (v < bv) { bv = v; bj = q * 256 + tid; }
        }
        pv[tid] = bv; pi[tid] = bj;
        __syncthreads();
        if (tid < 32) {
            float v = pv[tid]; int jj = pi[tid];
            for (int s = 1; s < 8; ++s) {
                float v2 = pv[tid + 32 * s]; int j2 = pi[tid + 32 * s];
                if (v2 < v || (v2 == v && j2 < jj)) { v = v2; jj = j2; }
            }
            for (int msk = 16; msk >= 1; msk >>= 1) {
                float v2 = __shfl_xor(v, msk, 32);
                int   j2 = __shfl_xor(jj, msk, 32);
                if (v2 < v || (v2 == v && j2 < jj)) { v = v2; jj = j2; }
            }
            if (tid == 0) {
                size_t o = (size_t)bi * KNB + k;
                eidx[o] = jj;
                dnb[o]  = v;          // adjusted value, as in reference D_neighbors
                eidx_out[o] = jj;
                sSelJ = jj;
            }
        }
        __syncthreads();
        int jw = sSelJ;
        if ((jw & 255) == tid) lv[jw >> 8] = __builtin_inff();   // owner invalidates
    }
}

// ---------------------------------------------------------------------------
// Kernel 4: fused feature build + WMMA edge GEMM + LayerNorm.
// One 128-thread (4-wave) block per (b,i): 48 edges x 416 feats -> 48 x 128.
// Each wave owns n-tiles {wave, wave+4}; per kt it loads 3 A frags (LDS) +
// 2 B frags (global, L2-resident) and issues 6 WMMAs, A reused across N.
// ---------------------------------------------------------------------------
__global__ __launch_bounds__(128) void edge_kernel(
    const float* __restrict__ X, const float* __restrict__ Cb,
    const int* __restrict__ R_idx, const int* __restrict__ chains,
    const float* __restrict__ W_pe, const float* __restrict__ b_pe,
    const _Float16* __restrict__ Wf, const float* __restrict__ ln_g,
    const float* __restrict__ ln_b, const int* __restrict__ eidx,
    const float* __restrict__ dnb, float* __restrict__ outE)
{
    __shared__ _Float16 sE[KNB * EIN_P];    // ~40.7 KB (A matrix, f16, padded rows)
    __shared__ float    sD25[KNB * 25];     // 4800 B
    __shared__ float    sOut[KNB * EOUT_P]; // ~25.3 KB (padded rows)
    __shared__ int      sDpe[KNB];
    __shared__ int      sNb[KNB];
    __shared__ float    sAtoms[5][3];

    const int tid = threadIdx.x;
    const int bi  = blockIdx.x;           // b*L + i
    const int b   = bi / L_DIM;
    const size_t rowBase = (size_t)bi * KNB;

    // ---- per-edge metadata + own atoms ----
    if (tid < KNB) {
        int j = eidx[rowBase + tid];
        sNb[tid] = j;
        int gj = b * L_DIM + j;
        int off  = R_idx[bi] - R_idx[gj];
        int same = (chains[bi] == chains[gj]) ? 1 : 0;
        int d = off + 32;
        d = d < 0 ? 0 : (d > 64 ? 64 : d);
        sDpe[tid] = same ? d : 65;
    }
    if (tid >= 64 && tid < 64 + 15) {
        int a = (tid - 64) / 3, dd = (tid - 64) % 3;
        sAtoms[a][dd] = (a == 4) ? Cb[(size_t)bi * 3 + dd]
                                 : X[((size_t)bi * 4 + a) * 3 + dd];
    }
    __syncthreads();

    // ---- 25 inter-atomic distances per edge ----
    for (int t = tid; t < KNB * 25; t += 128) {
        int e = t / 25, p = t - e * 25;
        if (p == 0) {
            sD25[t] = dnb[rowBase + e];
        } else {
            int pa = c_pairA[p - 1], pb = c_pairB[p - 1];
            size_t gj = (size_t)b * L_DIM + sNb[e];
            const float* q = (pb == 4) ? (Cb + gj * 3) : (X + (gj * 4 + pb) * 3);
            float dx = sAtoms[pa][0] - q[0];
            float dy = sAtoms[pa][1] - q[1];
            float dz = sAtoms[pa][2] - q[2];
            sD25[t] = sqrtf(dx * dx + dy * dy + dz * dz + 1e-6f);
        }
    }
    __syncthreads();

    // ---- build 416-dim feature rows (PE[16] | 25 x RBF[16]) as f16 ----
    for (int f = tid; f < KNB * EIN; f += 128) {
        int e = f / EIN;
        int c = f - e * EIN;
        float val;
        if (c < 16) {
            val = W_pe[sDpe[e] * 16 + c] + b_pe[c];
        } else {
            int blk = (c - 16) >> 4;
            int jj  = (c - 16) & 15;
            float D  = sD25[e * 25 + blk];
            float mu = 2.0f + (20.0f / 15.0f) * (float)jj;
            float z  = (D - mu) * (1.0f / 1.25f);
            val = __expf(-z * z);
        }
        sE[e * EIN_P + c] = (_Float16)val;
    }
    __syncthreads();

    // ---- WMMA GEMM: [48 x 416](f16) x [416 x 128](f16) -> [48 x 128](f32) ----
    const int wave = tid >> 5;
    const int lane = tid & 31;
    const int lrow = lane & 15;
    const int klo  = (lane & 16) ? 8 : 0;

    v8f acc[2][MTILES] = {};
    #pragma unroll 1
    for (int kt = 0; kt < KTILES; ++kt) {
        int cb = kt * 32 + klo;
        v16h a[MTILES];
        #pragma unroll
        for (int mt = 0; mt < MTILES; ++mt) {
            const _Float16* arow = &sE[(mt * 16 + lrow) * EIN_P + cb];
            union { v16h v; v8h h[2]; } ua;
            ua.h[0] = *(const v8h*)&arow[0];     // K = cb..cb+7
            ua.h[1] = *(const v8h*)&arow[16];    // K = cb+16..cb+23
            a[mt] = ua.v;
        }
        v16h bfr[2];
        #pragma unroll
        for (int h = 0; h < 2; ++h) {
            int nt = wave + 4 * h;
            const _Float16* wb = Wf + ((size_t)(nt * KTILES + kt) * 32 + lane) * 16;
            union { v16h v; v8h h2[2]; } ub;
            ub.h2[0] = *(const v8h*)&wb[0];
            ub.h2[1] = *(const v8h*)&wb[8];
            bfr[h] = ub.v;
        }
        #pragma unroll
        for (int h = 0; h < 2; ++h)
            #pragma unroll
            for (int mt = 0; mt < MTILES; ++mt)
                acc[h][mt] = __builtin_amdgcn_wmma_f32_16x16x32_f16(
                    false, a[mt], false, bfr[h], (short)0, acc[h][mt], false, false);
    }
    // C/D layout: VGPR r -> M = r (lanes 0-15) or r+8 (lanes 16-31), N = lane&15
    const int mhi = (lane & 16) ? 8 : 0;
    #pragma unroll
    for (int h = 0; h < 2; ++h) {
        int n = (wave + 4 * h) * 16 + lrow;
        #pragma unroll
        for (int mt = 0; mt < MTILES; ++mt)
            #pragma unroll
            for (int r = 0; r < 8; ++r)
                sOut[(mt * 16 + mhi + r) * EOUT_P + n] = acc[h][mt][r];
    }
    __syncthreads();

    // ---- LayerNorm over 128 features, one wave per edge ----
    for (int e = wave; e < KNB; e += 4) {
        float x[4], s = 0.0f, ss = 0.0f;
        #pragma unroll
        for (int q = 0; q < 4; ++q) {
            x[q] = sOut[e * EOUT_P + lane + 32 * q];
            s += x[q]; ss += x[q] * x[q];
        }
        for (int msk = 16; msk >= 1; msk >>= 1) {
            s  += __shfl_xor(s, msk, 32);
            ss += __shfl_xor(ss, msk, 32);
        }
        float mean = s * (1.0f / 128.0f);
        float var  = ss * (1.0f / 128.0f) - mean * mean;
        float rstd = rsqrtf(var + 1e-5f);
        size_t ob = (rowBase + e) * EOUT;
        #pragma unroll
        for (int q = 0; q < 4; ++q) {
            int c = lane + 32 * q;
            outE[ob + c] = (x[q] - mean) * rstd * ln_g[c] + ln_b[c];
        }
    }
}

// ---------------------------------------------------------------------------
extern "C" void kernel_launch(void* const* d_in, const int* in_sizes, int n_in,
                              void* d_out, int out_size, void* d_ws, size_t ws_size,
                              hipStream_t stream) {
    const float* X       = (const float*)d_in[0];
    const float* mask    = (const float*)d_in[1];
    const int*   R_idx   = (const int*)d_in[2];
    const int*   chains  = (const int*)d_in[3];
    const float* W_pe    = (const float*)d_in[4];
    const float* b_pe    = (const float*)d_in[5];
    const float* W_edge  = (const float*)d_in[6];
    const float* ln_g    = (const float*)d_in[7];
    const float* ln_b    = (const float*)d_in[8];

    const int BL = B_DIM * L_DIM;                         // 6144
    float* outE   = (float*)d_out;                        // [B,L,K,128]
    int*   outIdx = (int*)(outE + (size_t)BL * KNB * EOUT);

    // workspace layout
    char* ws = (char*)d_ws;
    float*    wsCb  = (float*)ws;                                   //  73728 B
    int*      wsIdx = (int*)(ws + 73728);                           // 1179648 B
    float*    wsDnb = (float*)(ws + 73728 + 1179648);               // 1179648 B
    _Float16* wsWf  = (_Float16*)(ws + 73728 + 2 * 1179648);        // 106496 B

    cb_kernel<<<(BL + 255) / 256, 256, 0, stream>>>(X, wsCb);
    wconv_kernel<<<(NTILES * KTILES * 32 + 255) / 256, 256, 0, stream>>>(W_edge, wsWf);
    knn_kernel<<<BL, 256, 0, stream>>>(X, mask, wsIdx, wsDnb, outIdx);
    edge_kernel<<<BL, 128, 0, stream>>>(X, wsCb, R_idx, chains, W_pe, b_pe,
                                        wsWf, ln_g, ln_b, wsIdx, wsDnb, outE);
}
